// EnhancedMultiTaskDecoders_40561671143603
// MI455X (gfx1250) — matrix operand order repaired and checked by reading, not compiled, not run on hardware
//
#include <hip/hip_runtime.h>
#include <hip/hip_bf16.h>

typedef __bf16 v16bf __attribute__((ext_vector_type(16)));
typedef __bf16 v8bf  __attribute__((ext_vector_type(8)));
typedef float  v8f   __attribute__((ext_vector_type(8)));

#define NROWS 262144
#define DIN   256

// ---------------------------------------------------------------------------
// Prep kernels
// ---------------------------------------------------------------------------
__global__ __launch_bounds__(64)
void zero_counts_kernel(int* counts) {
    if (threadIdx.x < 4) counts[threadIdx.x] = 0;
}

// Convert fp32 weight (K x Nw, row-major) into bf16 pre-swizzled WMMA
// B-fragment order: dst[((nt*ksteps + kk)*32 + lane)*16 + e]
// where n = nt*16 + (lane&15), k = kk*32 + 8*(lane>>4) + (e&7) + 16*(e>>3)
// (B layout matches A: lane half h holds K = {8h..8h+7, 16+8h..16+8h+7})
__global__ __launch_bounds__(256)
void swizzle_w_kernel(const float* __restrict__ src, __bf16* __restrict__ dst,
                      int Nw, int ksteps, int total) {
    int i = blockIdx.x * blockDim.x + threadIdx.x;
    if (i >= total) return;
    int e     = i & 15;
    int lane  = (i >> 4) & 31;
    int frame = i >> 9;
    int kk    = frame % ksteps;
    int nt    = frame / ksteps;
    int n = nt * 16 + (lane & 15);
    int k = kk * 32 + ((lane >> 4) << 3) + (e & 7) + ((e >> 3) << 4);
    dst[i] = (__bf16)src[k * Nw + n];
}

// Compact row indices per group (order within a group is irrelevant: each
// row's output depends only on its own data -> output is deterministic).
__global__ __launch_bounds__(256)
void build_index_kernel(const int* __restrict__ labels, int* __restrict__ counts,
                        int* __restrict__ idx, int n) {
    int i = blockIdx.x * blockDim.x + threadIdx.x;
    if (i < n) {
        int g = labels[i] & 3;
        int pos = atomicAdd(&counts[g], 1);
        idx[g * n + pos] = i;
    }
}

// ---------------------------------------------------------------------------
// Decoder: one wave handles one 16-row tile of one group.
// Layer1: 16x256 @ 256xH  (WMMA bf16, f32 acc)  -> LN -> ReLU
// Layer2: 16xH   @ HxH/2  (WMMA bf16, f32 acc)  -> LN -> ReLU
// Layer3: fused dot with W3 during LN2 apply.
// ---------------------------------------------------------------------------
template<int H>
__global__ __launch_bounds__(128)
void decode_tile_kernel(const float* __restrict__ x,
                        const int*   __restrict__ rowIdx,
                        const int*   __restrict__ cntPtr,
                        const __bf16* __restrict__ w1,   // swizzled B frags
                        const __bf16* __restrict__ w2,   // swizzled B frags
                        const float* __restrict__ b1, const float* __restrict__ g1,
                        const float* __restrict__ be1,
                        const float* __restrict__ b2, const float* __restrict__ g2,
                        const float* __restrict__ be2,
                        const float* __restrict__ w3, const float* __restrict__ b3,
                        float* __restrict__ out) {
    constexpr int H2  = H / 2;
    constexpr int NT1 = H / 16;        // layer-1 N tiles
    constexpr int K1  = DIN / 32;      // layer-1 K steps (=8)
    constexpr int NT2 = H2 / 16;       // layer-2 N tiles
    constexpr int K2  = H / 32;        // layer-2 K steps
    constexpr int SC  = H  + 4;        // f32 LDS row stride (bank stagger)
    constexpr int SA  = H  + 8;        // bf16 LDS row stride (16B aligned + stagger)
    constexpr int SC2 = H2 + 4;
    constexpr int SLAB = ((16 * SC * 4 + 16 * SA * 2) + 31) & ~31;

    __shared__ __align__(32) unsigned char smem[4 * SLAB];

    const int lane = threadIdx.x & 31;
    const int wv   = threadIdx.x >> 5;
    const int m    = lane & 15;        // row within tile (A/C layout)
    const int h    = lane >> 4;        // K-octet half (A) / M-half (C)

    float*  ldsC = (float*)(smem + wv * SLAB);
    __bf16* ldsA = (__bf16*)(smem + wv * SLAB + 16 * SC * 4);

    const int cnt      = *cntPtr;
    const int tile     = blockIdx.x * 4 + wv;
    const int rowStart = tile << 4;
    if (rowStart >= cnt) return;                   // waves independent: safe
    const int  myIdx = rowStart + m;
    const bool valid = myIdx < cnt;
    const int  ridx  = rowIdx[valid ? myIdx : (cnt - 1)];
    const float* xr  = x + (size_t)ridx * DIN;

    // ---------------- Layer 1 GEMM ----------------
    v8f c1[NT1] = {};
#pragma unroll
    for (int kk = 0; kk < K1; ++kk) {
        // A fragment: lane m+16h holds row m, K = 32kk + {8h..8h+7, 16+8h..16+8h+7}
        v8f q0 = *(const v8f*)(xr + kk * 32 + h * 8);
        v8f q1 = *(const v8f*)(xr + kk * 32 + 16 + h * 8);
        v16bf a;
#pragma unroll
        for (int e = 0; e < 8; ++e) { a[e] = (__bf16)q0[e]; a[e + 8] = (__bf16)q1[e]; }
#pragma unroll
        for (int nt = 0; nt < NT1; ++nt) {
            v16bf b = *(const v16bf*)(w1 + (((nt * K1 + kk) * 32 + lane) << 4));
            c1[nt] = __builtin_amdgcn_wmma_f32_16x16x32_bf16(
                false, a, false, b, (short)0, c1[nt], false, false);
        }
    }

    // C -> LDS (row-major, +bias). C layout: VGPR r = (M=r+8h, N=nt*16+m)
#pragma unroll
    for (int nt = 0; nt < NT1; ++nt) {
        float bb = b1[nt * 16 + m];
#pragma unroll
        for (int r = 0; r < 8; ++r)
            ldsC[(r + 8 * h) * SC + nt * 16 + m] = c1[nt][r] + bb;
    }
    asm volatile("s_wait_dscnt 0" ::: "memory");

    // ---------------- LN1 + ReLU -> bf16 A-fragments in LDS ----------------
    {
        const int Hh = H / 2;
        float s = 0.f, ss = 0.f;
        for (int j = 0; j < Hh; ++j) {
            float v = ldsC[m * SC + h * Hh + j];
            s += v; ss += v * v;
        }
        s  += __shfl_xor(s, 16, 32);
        ss += __shfl_xor(ss, 16, 32);
        float mu = s * (1.0f / H);
        float rs = rsqrtf(ss * (1.0f / H) - mu * mu + 1e-5f);
        for (int j = 0; j < Hh; ++j) {
            int nc = h * Hh + j;
            float v = (ldsC[m * SC + nc] - mu) * rs * g1[nc] + be1[nc];
            ldsA[m * SA + nc] = (__bf16)fmaxf(v, 0.f);
        }
    }
    asm volatile("s_wait_dscnt 0" ::: "memory");

    // ---------------- Layer 2 GEMM ----------------
    v8f c2[NT2] = {};
#pragma unroll
    for (int kk = 0; kk < K2; ++kk) {
        v8bf a0 = *(const v8bf*)(ldsA + m * SA + kk * 32 + h * 8);
        v8bf a1 = *(const v8bf*)(ldsA + m * SA + kk * 32 + 16 + h * 8);
        v16bf a = __builtin_shufflevector(a0, a1, 0, 1, 2, 3, 4, 5, 6, 7,
                                          8, 9, 10, 11, 12, 13, 14, 15);
#pragma unroll
        for (int nt = 0; nt < NT2; ++nt) {
            v16bf b = *(const v16bf*)(w2 + (((nt * K2 + kk) * 32 + lane) << 4));
            c2[nt] = __builtin_amdgcn_wmma_f32_16x16x32_bf16(
                false, a, false, b, (short)0, c2[nt], false, false);
        }
    }

#pragma unroll
    for (int nt = 0; nt < NT2; ++nt) {
        float bb = b2[nt * 16 + m];
#pragma unroll
        for (int r = 0; r < 8; ++r)
            ldsC[(r + 8 * h) * SC2 + nt * 16 + m] = c2[nt][r] + bb;
    }
    asm volatile("s_wait_dscnt 0" ::: "memory");

    // ---------------- LN2 + ReLU + fused dot(W3) ----------------
    {
        const int Hh = H2 / 2;
        float s = 0.f, ss = 0.f;
        for (int j = 0; j < Hh; ++j) {
            float v = ldsC[m * SC2 + h * Hh + j];
            s += v; ss += v * v;
        }
        s  += __shfl_xor(s, 16, 32);
        ss += __shfl_xor(ss, 16, 32);
        float mu = s * (1.0f / H2);
        float rs = rsqrtf(ss * (1.0f / H2) - mu * mu + 1e-5f);
        float dot = 0.f;
        for (int j = 0; j < Hh; ++j) {
            int nc = h * Hh + j;
            float v = (ldsC[m * SC2 + nc] - mu) * rs * g2[nc] + be2[nc];
            dot += fmaxf(v, 0.f) * w3[nc];
        }
        dot += __shfl_xor(dot, 16, 32);
        if (h == 0 && valid) out[ridx] = dot + b3[0];
    }
}

// ---------------------------------------------------------------------------
// Host launcher
// ---------------------------------------------------------------------------
extern "C" void kernel_launch(void* const* d_in, const int* in_sizes, int n_in,
                              void* d_out, int out_size, void* d_ws, size_t ws_size,
                              hipStream_t stream) {
    (void)in_sizes; (void)n_in; (void)out_size; (void)ws_size;
    const int Nn = NROWS;
    const float* x      = (const float*)d_in[0];
    const int*   labels = (const int*)d_in[1];
    // params tuple order: W1,b1,g1,be1,W2,b2,g2,be2,W3,b3 per group
    auto P = [&](int g, int j) { return (const float*)d_in[2 + 10 * g + j]; };
    const int Hs[4] = {64, 64, 128, 128};

    char* ws = (char*)d_ws;
    int* counts = (int*)ws;
    int* idx    = (int*)(ws + 256);
    __bf16* wbf = (__bf16*)(ws + 256 + (size_t)4 * Nn * sizeof(int));
    __bf16* w1p[4]; __bf16* w2p[4];
    size_t off = 0;
    for (int g = 0; g < 4; ++g) { w1p[g] = wbf + off; off += (size_t)DIN * Hs[g]; }
    for (int g = 0; g < 4; ++g) { w2p[g] = wbf + off; off += (size_t)Hs[g] * (Hs[g] / 2); }

    zero_counts_kernel<<<1, 64, 0, stream>>>(counts);
    for (int g = 0; g < 4; ++g) {
        int tot1 = DIN * Hs[g];
        swizzle_w_kernel<<<(tot1 + 255) / 256, 256, 0, stream>>>(
            P(g, 0), w1p[g], Hs[g], DIN / 32, tot1);
        int tot2 = Hs[g] * (Hs[g] / 2);
        swizzle_w_kernel<<<(tot2 + 255) / 256, 256, 0, stream>>>(
            P(g, 4), w2p[g], Hs[g] / 2, Hs[g] / 32, tot2);
    }
    build_index_kernel<<<Nn / 256, 256, 0, stream>>>(labels, counts, idx, Nn);

    const int maxTiles = (Nn + 15) / 16;        // worst case: all rows one group
    const int blocks   = (maxTiles + 3) / 4;    // 4 waves (tiles) per block
    for (int g = 0; g < 4; ++g) {
        if (Hs[g] == 64) {
            decode_tile_kernel<64><<<blocks, 128, 0, stream>>>(
                x, idx + (size_t)g * Nn, counts + g, w1p[g], w2p[g],
                P(g, 1), P(g, 2), P(g, 3), P(g, 5), P(g, 6), P(g, 7),
                P(g, 8), P(g, 9), (float*)d_out);
        } else {
            decode_tile_kernel<128><<<blocks, 128, 0, stream>>>(
                x, idx + (size_t)g * Nn, counts + g, w1p[g], w2p[g],
                P(g, 1), P(g, 2), P(g, 3), P(g, 5), P(g, 6), P(g, 7),
                P(g, 8), P(g, 9), (float*)d_out);
        }
    }
}
